// NNBase_22539988369680
// MI455X (gfx1250) — matrix operand order repaired and checked by reading, not compiled, not run on hardware
//
#include <hip/hip_runtime.h>
#include <math.h>

// Problem constants (hardcoded in reference)
#define CCH   32            // hidden channels
#define PIX   49            // 7x7 spatial
#define NA    8             // actions
#define TT    64            // time steps
#define NBATCH 128          // batch
#define HH    1568          // CCH*PIX
#define K2    288           // GEMM K = CCH*9 taps (action folded into epilogue)
#define KKC   9             // K chunks of 32 for wmma_f32_16x16x32_f16
#define NT    256           // threads per block = 8 waves (wave32)

typedef __attribute__((ext_vector_type(16))) _Float16 v16h;
typedef __attribute__((ext_vector_type(8)))  float    v8f;

// ---------------- dynamic LDS layout (bytes, all 16B aligned) ----------------
#define SZ_WCONV   (KKC*2*512*2)          // one conv's swizzled B fragments: 18432 B
#define LDS_WGT    0
#define SZ_WGT     (6*SZ_WCONV)           // 110592
#define LDS_IMA    (LDS_WGT + SZ_WGT)
#define SZ_IM      (64*K2*2)              // 36864 (rows 49..63 stay zero)
#define LDS_IMB    (LDS_IMA + SZ_IM)
#define LDS_H      (LDS_IMB + SZ_IM)      // persistent hidden, f32
#define LDS_PREV   (LDS_H + HH*4)         // masked hidden, f32
#define LDS_PREV16 (LDS_PREV + HH*4)      // masked hidden, f16
#define LDS_MID    (LDS_PREV16 + HH*2)    // relu(conv1) f16
#define LDS_PREVR  (LDS_MID + HH*2)       // prev*reset f16
#define LDS_UPD    (LDS_PREVR + HH*2)     // update gate f32
#define LDS_PMU    (LDS_UPD + HH*4)       // p_mu f32
#define LDS_SW     (LDS_PMU + HH*4)       // action-conv scalar terms [3][32][9] f32
#define LDS_ACT    (LDS_SW + 3*CCH*9*4)   // masked one-hot action [8] f32
#define LDS_TOTAL  (LDS_ACT + 64)         // 222336 bytes < 320KB/WGP

// im2col gather: dst[row=pixel][k = c*9+tap], rows 49..63 pre-zeroed once.
__device__ __forceinline__ void build_im2col(_Float16* dst, const _Float16* src, int tid) {
  for (int idx = tid; idx < PIX * K2; idx += NT) {
    int r   = idx / K2;
    int k   = idx - r * K2;
    int c   = k / 9, tap = k - c * 9;
    int y   = r / 7 + tap / 3 - 1;
    int xx  = (r % 7) + (tap % 3) - 1;
    _Float16 v = (_Float16)0.f;
    if ((unsigned)y < 7u && (unsigned)xx < 7u) v = src[c * PIX + y * 7 + xx];
    dst[r * K2 + k] = v;
  }
}

// One 16x16 output tile of a 64x32 = (64xK2)x(K2x32) GEMM.
// A fragment (16-bit A 16x32): lanes 0-15 hold M=lane, K = kb+{0..7,16..23};
// lanes 16-31 hold M=lane-16, K = kb+{8..15,24..31}. Row-major im2col makes each
// half a contiguous 16B chunk -> two ds_load_b128 per fragment.
// B fragments are pre-swizzled in LDS: per (kk,ni) tile, lane L owns 16 contiguous
// halves (n = ni*16+(L&15); K = kk*32 + (L<16?0:8) + j + (j<8?0:8)).
// Software-pipelined: fragments for kk+1 are loaded before consuming kk's, so the
// scheduler can overlap 4 outstanding ds_load_b128 with each v_wmma (partial
// s_wait_dscnt instead of dscnt==0 before every WMMA).
__device__ __forceinline__ v8f conv_gemm_tile(const _Float16* im, const _Float16* wB,
                                              int lane, int mi, int ni) {
  const int m    = mi * 16 + (lane & 15);
  const int krow = (lane & 16) ? 8 : 0;
  const _Float16* arow = im + m * K2 + krow;
  const _Float16* brow = wB + ni * 512 + lane * 16;   // +1024 halves per kk step

  union F { float4 q[2]; v16h v; };
  F a0, b0, a1, b1;
  {
    const float4* ap = reinterpret_cast<const float4*>(arow);
    const float4* bp = reinterpret_cast<const float4*>(brow);
    a0.q[0] = ap[0]; a0.q[1] = ap[2];
    b0.q[0] = bp[0]; b0.q[1] = bp[1];
  }
  v8f acc = {0.f, 0.f, 0.f, 0.f, 0.f, 0.f, 0.f, 0.f};
#pragma unroll
  for (int kk = 0; kk < KKC; ++kk) {
    if (kk + 1 < KKC) {
      const float4* apn = reinterpret_cast<const float4*>(arow + (kk + 1) * 32);
      const float4* bpn = reinterpret_cast<const float4*>(brow + (kk + 1) * 1024);
      a1.q[0] = apn[0]; a1.q[1] = apn[2];
      b1.q[0] = bpn[0]; b1.q[1] = bpn[1];
    }
    acc = __builtin_amdgcn_wmma_f32_16x16x32_f16(false, a0.v, false, b0.v,
                                                 (short)0, acc, false, false);
    a0 = a1; b0 = b1;
  }
  return acc;
}

__global__ __launch_bounds__(NT, 1)
void convgru_wmma_kernel(const float* __restrict__ x,   const float* __restrict__ hxs,
                         const float* __restrict__ att, const float* __restrict__ masks,
                         const float* __restrict__ pa,
                         const float* __restrict__ Wr1, const float* __restrict__ br1,
                         const float* __restrict__ Wr2, const float* __restrict__ br2,
                         const float* __restrict__ Wu1, const float* __restrict__ bu1,
                         const float* __restrict__ Wu2, const float* __restrict__ bu2,
                         const float* __restrict__ Wo1, const float* __restrict__ bo1,
                         const float* __restrict__ Wo2, const float* __restrict__ bo2,
                         float* __restrict__ out) {
  extern __shared__ char smem[];
  _Float16* wgt    = (_Float16*)(smem + LDS_WGT);
  _Float16* imA    = (_Float16*)(smem + LDS_IMA);
  _Float16* imB    = (_Float16*)(smem + LDS_IMB);
  float*    hbuf   = (float*)(smem + LDS_H);
  float*    prev   = (float*)(smem + LDS_PREV);
  _Float16* prev16 = (_Float16*)(smem + LDS_PREV16);
  _Float16* mid    = (_Float16*)(smem + LDS_MID);
  _Float16* prevr  = (_Float16*)(smem + LDS_PREVR);
  float*    upd    = (float*)(smem + LDS_UPD);
  float*    pmu    = (float*)(smem + LDS_PMU);
  float*    sw     = (float*)(smem + LDS_SW);
  float*    actv   = (float*)(smem + LDS_ACT);

  const int n     = blockIdx.x;           // one sample's whole T-chain per block
  const int tid   = threadIdx.x;
  const int lane  = tid & 31;
  const int wv    = tid >> 5;             // 8 waves -> 8 output tiles (4 M x 2 N)
  const int mi    = wv >> 1;
  const int ni    = wv & 1;
  const int co    = ni * 16 + (lane & 15);             // D col = out channel
  const int pbase = mi * 16 + ((lane & 16) ? 8 : 0);   // D row base = pixel

  const float* W1p[3] = {Wu1, Wr1, Wo1};  // gate order: update, reset, out
  const float* W2p[3] = {Wu2, Wr2, Wo2};
  const float* b1p[3] = {bu1, br1, bo1};
  const float* b2p[3] = {bu2, br2, bo2};

  // ---- one-time: swizzle conv weights (f32 OIHW -> f16 WMMA-B fragments) ----
  for (int idx = tid; idx < 6 * KKC * 2 * 512; idx += NT) {
    int j   = idx & 15;
    int ln  = (idx >> 4) & 31;
    int rst = idx >> 9;
    int nib = rst & 1;  rst >>= 1;
    int kk  = rst % KKC;
    int c6  = rst / KKC;                      // 0..5 = u1,u2,r1,r2,o1,o2
    int k   = kk * 32 + ((ln & 16) ? 8 : 0) + (j < 8 ? j : j + 8);
    int nn  = nib * 16 + (ln & 15);
    int c   = k / 9, tap = k % 9;
    const float* Wsrc = (c6 & 1) ? W2p[c6 >> 1] : W1p[c6 >> 1];
    float wvv = (c6 & 1) ? Wsrc[nn * (CCH * 9) + c * 9 + tap]
                         : Wsrc[nn * ((NA + CCH) * 9) + (NA + c) * 9 + tap];
    wgt[((c6 * KKC + kk) * 2 + nib) * 512 + ln * 16 + j] = (_Float16)wvv;
  }
  // zero both im2col buffers once (pad rows 49..63 never rewritten)
  for (int idx = tid; idx < 2 * 64 * K2; idx += NT) imA[idx] = (_Float16)0.f;
  // load initial hidden state
  for (int i = tid; i < HH; i += NT) hbuf[i] = hxs[(size_t)n * HH + i];
  __syncthreads();

  for (int t = 0; t < TT; ++t) {
    const size_t row = (size_t)t * NBATCH + n;
    const float  m   = masks[row];
    const float* xr  = x + row * HH;
    const float* ar  = att + row * HH;

    // one global_prefetch_b8 per 128B line: attention row (used ~10 phases later)
    // and next step's x row
    for (int off = tid * 32; off < HH; off += NT * 32) {
      __builtin_prefetch(ar + off, 0, 3);
      if (t + 1 < TT) __builtin_prefetch(xr + NBATCH * HH + off, 0, 3);
    }

    // phase A: hidden = h*m + x*(1-m); stage f32 + f16 copies
    for (int i = tid; i < HH; i += NT) {
      float hv = hbuf[i] * m + xr[i] * (1.f - m);
      prev[i]   = hv;
      prev16[i] = (_Float16)hv;
    }
    if (tid < NA) actv[tid] = pa[row * NA + tid] * m;
    __syncthreads();

    // action channels are spatially constant: fold their conv into per-tap scalars
    for (int idx = tid; idx < 3 * CCH * 9; idx += NT) {
      int g  = idx / (CCH * 9);
      int r2 = idx - g * (CCH * 9);
      int cc = r2 / 9, tap = r2 - cc * 9;
      const float* W1 = W1p[g];
      float s = 0.f;
      for (int a = 0; a < NA; ++a) s += W1[cc * ((NA + CCH) * 9) + a * 9 + tap] * actv[a];
      sw[idx] = s;
    }
    build_im2col(imA, prev16, tid);       // shared by update & reset conv1
    __syncthreads();

    for (int g = 0; g < 3; ++g) {
      if (g == 2) { build_im2col(imA, prevr, tid); __syncthreads(); }  // out gate input

      // conv1: 9x v_wmma, epilogue adds bias + action term, ReLU -> mid (f16)
      {
        v8f acc = conv_gemm_tile(imA, wgt + (2 * g) * (KKC * 2 * 512), lane, mi, ni);
        float b = b1p[g][co];
        const float* swg = sw + g * (CCH * 9) + co * 9;
#pragma unroll
        for (int r2 = 0; r2 < 8; ++r2) {
          int p = pbase + r2;
          if (p < PIX) {
            int y = p / 7, xx = p % 7;
            float v = acc[r2] + b;
            for (int dy = -1; dy <= 1; ++dy)
              for (int dx = -1; dx <= 1; ++dx) {
                int yy = y + dy, xc = xx + dx;
                if ((unsigned)yy < 7u && (unsigned)xc < 7u)
                  v += swg[(dy + 1) * 3 + (dx + 1)];
              }
            v = v > 0.f ? v : 0.f;
            mid[co * PIX + p] = (_Float16)v;
          }
        }
      }
      __syncthreads();
      build_im2col(imB, mid, tid);
      __syncthreads();

      // conv2 + gate nonlinearity
      {
        v8f acc = conv_gemm_tile(imB, wgt + (2 * g + 1) * (KKC * 2 * 512), lane, mi, ni);
        float b = b2p[g][co];
#pragma unroll
        for (int r2 = 0; r2 < 8; ++r2) {
          int p = pbase + r2;
          if (p < PIX) {
            float v  = acc[r2] + b;
            int   ix = co * PIX + p;
            if (g == 0) {
              upd[ix] = 1.f / (1.f + __expf(-v));                       // update
            } else if (g == 1) {
              float rr = 1.f / (1.f + __expf(-v));                      // reset
              prevr[ix] = (_Float16)(prev[ix] * rr);
            } else {
              float o = tanhf(v);                                       // candidate
              float u = upd[ix];
              pmu[ix] = prev[ix] * (1.f - u) + o * u;
            }
          }
        }
      }
      __syncthreads();
    }

    // combine: attended = x*att + p_mu*(1-att); write both outputs; carry hidden
    for (int i = tid; i < HH; i += NT) {
      float pm  = pmu[i];
      float a   = ar[i];
      float atd = xr[i] * a + pm * (1.f - a);
      out[row * HH + i] = pm;                                   // p_mu
      out[(size_t)TT * NBATCH * HH + row * HH + i] = atd;       // attended
      hbuf[i] = atd;
    }
    __syncthreads();
  }

  // final hidden state hxs_f
  for (int i = tid; i < HH; i += NT)
    out[(size_t)2 * TT * NBATCH * HH + (size_t)n * HH + i] = hbuf[i];
}

extern "C" void kernel_launch(void* const* d_in, const int* in_sizes, int n_in,
                              void* d_out, int out_size, void* d_ws, size_t ws_size,
                              hipStream_t stream) {
  (void)in_sizes; (void)n_in; (void)out_size; (void)d_ws; (void)ws_size;
  const float* x    = (const float*)d_in[0];
  const float* hxs  = (const float*)d_in[1];
  const float* att  = (const float*)d_in[2];
  const float* mask = (const float*)d_in[3];
  const float* pa   = (const float*)d_in[4];
  const float* Wr1  = (const float*)d_in[5];
  const float* br1  = (const float*)d_in[6];
  const float* Wr2  = (const float*)d_in[7];
  const float* br2  = (const float*)d_in[8];
  const float* Wu1  = (const float*)d_in[9];
  const float* bu1  = (const float*)d_in[10];
  const float* Wu2  = (const float*)d_in[11];
  const float* bu2  = (const float*)d_in[12];
  const float* Wo1  = (const float*)d_in[13];
  const float* bo1  = (const float*)d_in[14];
  const float* Wo2  = (const float*)d_in[15];
  const float* bo2  = (const float*)d_in[16];
  float* out = (float*)d_out;

  convgru_wmma_kernel<<<NBATCH, NT, LDS_TOTAL, stream>>>(
      x, hxs, att, mask, pa,
      Wr1, br1, Wr2, br2, Wu1, bu1, Wu2, bu2, Wo1, bo1, Wo2, bo2, out);
}
